// GCN_39702677684583
// MI455X (gfx1250) — matrix operand order repaired and checked by reading, not compiled, not run on hardware
//
#include <hip/hip_runtime.h>
#include <hip/hip_bf16.h>

typedef __attribute__((ext_vector_type(16))) _Float16 v16h;
typedef __attribute__((ext_vector_type(8)))  float    v8f;

#define N_NODES  100000
#define N_EDGES  800000
#define N_GRAPHS 512
#define N_CLASS  10

// ---------------- degree / dinv ----------------
__global__ void k_init_one(float* p, int n) {
    int i = blockIdx.x * blockDim.x + threadIdx.x;
    if (i < n) p[i] = 1.0f;
}
__global__ void k_zero(float* p, int n) {
    int i = blockIdx.x * blockDim.x + threadIdx.x;
    if (i < n) p[i] = 0.0f;
}
__global__ void k_deg_accum(const long long* __restrict__ dst, float* __restrict__ deg, int nE) {
    int e = blockIdx.x * blockDim.x + threadIdx.x;
    if (e < nE) atomicAdd(&deg[(int)dst[e]], 1.0f);
}
__global__ void k_rsqrt_inplace(float* p, int n) {
    int i = blockIdx.x * blockDim.x + threadIdx.x;
    if (i < n) p[i] = rsqrtf(p[i]);
}

// ---------------- WMMA GEMM:  P = Acc = (X @ W) * dinv[row] ----------------
// One wave computes a 16x64 strip of the output (4 WMMA tiles, K unrolled by 32).
// Fragment layouts used by this kernel (wave32, A/B in contiguous k order):
//   A (f16 16x32): lane m=L&15, hi=L>>4;  a[j]   = X[m][16*hi + j]      j=0..7
//                                         a[j+8] = X[m][16*hi + 8 + j]
//   B (f16 32x16): lane n=L&15, hi=L>>4;  b[j]   = W[16*hi + j][n]
//   C (f32 16x16): lane n=L&15, hi=L>>4;  vgpr v = C[v + 8*hi][n]
template <int K, int C>
__global__ void k_gemm_wmma(const float* __restrict__ X, const float* __restrict__ W,
                            const float* __restrict__ dinv,
                            float* __restrict__ P, float* __restrict__ Acc, int N) {
    const int lane = threadIdx.x & 31;
    const int wv   = threadIdx.x >> 5;
    const int gw   = blockIdx.x * (blockDim.x >> 5) + wv;
    const int tilesN = C >> 6;                 // 64-wide strips
    const int tilesM = N >> 4;
    if (gw >= tilesM * tilesN) return;         // wave-uniform: EXEC stays full for WMMA
    const int tm = gw / tilesN, tn = gw % tilesN;
    const int row0 = tm << 4, n0 = tn << 6;
    const int m  = lane & 15;
    const int hi = lane >> 4;

    const float* __restrict__ xr = X + (size_t)(row0 + m) * K;
    v8f acc[4] = {v8f{}, v8f{}, v8f{}, v8f{}};

    #pragma unroll
    for (int k0 = 0; k0 < K; k0 += 32) {
        v16h a;
        #pragma unroll
        for (int j = 0; j < 8; ++j) {
            a[j]     = (_Float16)xr[k0 + 16 * hi + j];
            a[j + 8] = (_Float16)xr[k0 + 16 * hi + 8 + j];
        }
        #pragma unroll
        for (int t = 0; t < 4; ++t) {
            const int gn = n0 + (t << 4) + m;
            const float* __restrict__ wc = W + (size_t)(k0 + (hi << 4)) * C + gn;
            v16h b;
            #pragma unroll
            for (int j = 0; j < 16; ++j) b[j] = (_Float16)wc[(size_t)j * C];
            acc[t] = __builtin_amdgcn_wmma_f32_16x16x32_f16(
                false, a, false, b, (short)0, acc[t], false, false);
        }
    }

    #pragma unroll
    for (int v = 0; v < 8; ++v) {
        const int row = row0 + v + (hi << 3);
        const float dv = dinv[row];
        #pragma unroll
        for (int t = 0; t < 4; ++t) {
            const int col = n0 + (t << 4) + m;
            const float val = acc[t][v] * dv;
            P[(size_t)row * C + col]   = val;   // gathered by edges
            Acc[(size_t)row * C + col] = val;   // self-loop term pre-seeded
        }
    }
}

// ---------------- edge scatter:  Acc[dst] += P[src]  (L2-resident atomics) ----------------
template <int C>
__global__ void k_edge_scatter(const long long* __restrict__ src, const long long* __restrict__ dst,
                               const float* __restrict__ P, float* __restrict__ Acc, int nE) {
    constexpr int CH = C >> 2;                 // float4 chunks per row
    int tid = blockIdx.x * blockDim.x + threadIdx.x;
    if (tid >= nE * CH) return;
    const int e  = tid / CH;
    const int c4 = (tid % CH) << 2;
    const int s = (int)src[e], d = (int)dst[e];
    const float4 v = *(const float4*)(P + (size_t)s * C + c4);
    float* o = Acc + (size_t)d * C + c4;
    atomicAdd(o + 0, v.x);
    atomicAdd(o + 1, v.y);
    atomicAdd(o + 2, v.z);
    atomicAdd(o + 3, v.w);
}

// ---------------- epilogue:  Xout = act(dinv * Acc + b) ----------------
template <int C, bool RELU>
__global__ void k_epilogue(const float* __restrict__ Acc, const float* __restrict__ dinv,
                           const float* __restrict__ bias, float* __restrict__ Xout, int N) {
    constexpr int CH = C >> 2;
    int tid = blockIdx.x * blockDim.x + threadIdx.x;
    if (tid >= N * CH) return;
    const int i  = tid / CH;
    const int c4 = (tid % CH) << 2;
    const float dv = dinv[i];
    const float4 a = *(const float4*)(Acc + (size_t)i * C + c4);
    const float4 b = *(const float4*)(bias + c4);
    float4 r;
    r.x = dv * a.x + b.x; r.y = dv * a.y + b.y;
    r.z = dv * a.z + b.z; r.w = dv * a.w + b.w;
    if (RELU) {
        r.x = fmaxf(r.x, 0.f); r.y = fmaxf(r.y, 0.f);
        r.z = fmaxf(r.z, 0.f); r.w = fmaxf(r.w, 0.f);
    }
    *(float4*)(Xout + (size_t)i * C + c4) = r;
}

// ---------------- pooling + final linear ----------------
__global__ void k_pool_sum(const float* __restrict__ H, const long long* __restrict__ batch,
                           float* __restrict__ pooled, int N) {
    constexpr int C = 256, CH = 64;
    int tid = blockIdx.x * blockDim.x + threadIdx.x;
    if (tid >= N * CH) return;
    const int i  = tid / CH;
    const int c4 = (tid % CH) << 2;
    const int g = (int)batch[i];
    const float4 v = *(const float4*)(H + (size_t)i * C + c4);
    float* o = pooled + (size_t)g * C + c4;
    atomicAdd(o + 0, v.x);
    atomicAdd(o + 1, v.y);
    atomicAdd(o + 2, v.z);
    atomicAdd(o + 3, v.w);
}
__global__ void k_pool_count(const long long* __restrict__ batch, float* __restrict__ counts, int N) {
    int i = blockIdx.x * blockDim.x + threadIdx.x;
    if (i < N) atomicAdd(&counts[(int)batch[i]], 1.0f);
}
__global__ void k_final(const float* __restrict__ pooled, const float* __restrict__ counts,
                        const float* __restrict__ Wl, const float* __restrict__ bl,
                        float* __restrict__ out) {
    int tid = blockIdx.x * blockDim.x + threadIdx.x;
    if (tid >= N_GRAPHS * N_CLASS) return;
    const int g = tid / N_CLASS, c = tid % N_CLASS;
    const float inv = 1.0f / fmaxf(counts[g], 1.0f);
    float s = 0.f;
    #pragma unroll 4
    for (int f = 0; f < 256; ++f) s += pooled[(size_t)g * 256 + f] * Wl[(size_t)f * N_CLASS + c];
    out[tid] = s * inv + bl[c];
}

static inline int cdiv(long long a, long long b) { return (int)((a + b - 1) / b); }

extern "C" void kernel_launch(void* const* d_in, const int* in_sizes, int n_in,
                              void* d_out, int out_size, void* d_ws, size_t ws_size,
                              hipStream_t stream) {
    (void)in_sizes; (void)n_in; (void)out_size; (void)ws_size;
    const float*     x     = (const float*)d_in[0];
    const long long* ei    = (const long long*)d_in[1];   // int64 [2, E]
    const long long* batch = (const long long*)d_in[2];   // int64 [N]
    const float* W1 = (const float*)d_in[3];  const float* b1 = (const float*)d_in[4];
    const float* W2 = (const float*)d_in[5];  const float* b2 = (const float*)d_in[6];
    const float* W3 = (const float*)d_in[7];  const float* b3 = (const float*)d_in[8];
    const float* Wl = (const float*)d_in[9];  const float* bl = (const float*)d_in[10];
    float* out = (float*)d_out;

    const long long* src = ei;
    const long long* dst = ei + N_EDGES;

    // workspace layout (floats): needs ~295 MiB
    float* ws     = (float*)d_ws;
    float* dinv   = ws;                       // 100000 (padded to 100352)
    float* pooled = ws + 100352;              // 512*256 = 131072
    float* counts = pooled + 131072;          // 512
    float* B0     = ws + 231936;              // 100000*256
    float* B1     = B0 + 25600000;
    float* B2     = B1 + 25600000;

    const int T = 256;

    // deg -> dinv (in place)
    k_init_one<<<cdiv(N_NODES, T), T, 0, stream>>>(dinv, N_NODES);
    k_deg_accum<<<cdiv(N_EDGES, T), T, 0, stream>>>(dst, dinv, N_EDGES);
    k_rsqrt_inplace<<<cdiv(N_NODES, T), T, 0, stream>>>(dinv, N_NODES);

    const int tilesM = N_NODES / 16;          // 6250
    auto gemm_blocks = [&](int Cc) { return cdiv((long long)tilesM * (Cc / 64) * 32, T); };

    // Layer 1: 32 -> 64, ReLU
    k_gemm_wmma<32, 64><<<gemm_blocks(64), T, 0, stream>>>(x, W1, dinv, B0, B1, N_NODES);
    k_edge_scatter<64><<<cdiv((long long)N_EDGES * 16, T), T, 0, stream>>>(src, dst, B0, B1, N_EDGES);
    k_epilogue<64, true><<<cdiv((long long)N_NODES * 16, T), T, 0, stream>>>(B1, dinv, b1, B0, N_NODES);

    // Layer 2: 64 -> 128, ReLU
    k_gemm_wmma<64, 128><<<gemm_blocks(128), T, 0, stream>>>(B0, W2, dinv, B1, B2, N_NODES);
    k_edge_scatter<128><<<cdiv((long long)N_EDGES * 32, T), T, 0, stream>>>(src, dst, B1, B2, N_EDGES);
    k_epilogue<128, true><<<cdiv((long long)N_NODES * 32, T), T, 0, stream>>>(B2, dinv, b2, B1, N_NODES);

    // Layer 3: 128 -> 256, no ReLU
    k_gemm_wmma<128, 256><<<gemm_blocks(256), T, 0, stream>>>(B1, W3, dinv, B2, B0, N_NODES);
    k_edge_scatter<256><<<cdiv((long long)N_EDGES * 64, T), T, 0, stream>>>(src, dst, B2, B0, N_EDGES);
    k_epilogue<256, false><<<cdiv((long long)N_NODES * 64, T), T, 0, stream>>>(B0, dinv, b3, B2, N_NODES);

    // global mean pool + linear
    k_zero<<<cdiv(131072 + 512, T), T, 0, stream>>>(pooled, 131072 + 512);  // pooled + counts contiguous
    k_pool_sum<<<cdiv((long long)N_NODES * 64, T), T, 0, stream>>>(B2, batch, pooled, N_NODES);
    k_pool_count<<<cdiv(N_NODES, T), T, 0, stream>>>(batch, counts, N_NODES);
    k_final<<<cdiv(N_GRAPHS * N_CLASS, T), T, 0, stream>>>(pooled, counts, Wl, bl, out);
}
